// BertSelfAttentionMem_61770219651174
// MI455X (gfx1250) — matrix operand order repaired and checked by reading, not compiled, not run on hardware
//
#include <hip/hip_runtime.h>

typedef __attribute__((ext_vector_type(16))) _Float16 v16h;
typedef __attribute__((ext_vector_type(8)))  float    v8f;
typedef __attribute__((ext_vector_type(4)))  unsigned v4u;
typedef __attribute__((ext_vector_type(8)))  int      v8i;
typedef __attribute__((ext_vector_type(4)))  int      v4i;

#define BATCH   2
#define SEQ     2048
#define HID     1024
#define NHEAD   16
#define HDIM    64
#define TOKENS  4096      // BATCH*SEQ
#define N3      3072      // q|k|v output features

// workspace layout (bytes)
#define OFF_XH  0ull                       // 4096*1024 f16      = 8 MB
#define OFF_WP  (OFF_XH + 8388608ull)      // W panels f16       = 6 MB
#define OFF_QH  (OFF_WP + 6291456ull)      // [B,H,S,D] f16      = 8 MB
#define OFF_KH  (OFF_QH + 8388608ull)      // [B,H,S,D] f16      = 8 MB
#define OFF_VT  (OFF_KH + 8388608ull)      // [B,H,D,S] f16      = 8 MB

__device__ __forceinline__ v8f wmma32(v16h a, v16h b, v8f c) {
  // D = A(16x32 f16) * B(32x16 f16) + C(16x16 f32)
  return __builtin_amdgcn_wmma_f32_16x16x32_f16(
      false, a, false, b, (short)0, c, false, false);
}

__device__ __forceinline__ v16h load_frag2(const _Float16* p0, const _Float16* p1) {
  v16h t;
  *(uint4*)&t       = *(const uint4*)p0;
  *((uint4*)&t + 1) = *(const uint4*)p1;
  return t;
}

// CDNA5 async global->LDS copy: 16 bytes per lane, tracked by ASYNCcnt.
__device__ __forceinline__ void async_copy16(unsigned lds_off, const void* gptr) {
  asm volatile("global_load_async_to_lds_b128 %0, %1, off"
               :: "v"(lds_off), "v"(gptr) : "memory");
}
__device__ __forceinline__ void wait_async_le8() {
  asm volatile("s_wait_asynccnt 0x8" ::: "memory");
}
__device__ __forceinline__ void wait_async_0() {
  asm volatile("s_wait_asynccnt 0x0" ::: "memory");
}

// CDNA5 Tensor Data Mover: one 2D tile (tile_dim0 x tile_dim1, 2-byte elems)
// from global (row stride = stride0 elems) into LDS.  D# per ISA 8.3/8.4.
__device__ __forceinline__ void tdm_load_2d(unsigned lds_addr, const void* gptr,
                                            unsigned tensor_d0, unsigned tensor_d1,
                                            unsigned tile_d0, unsigned tile_d1,
                                            unsigned long long stride0) {
  unsigned long long ga = (unsigned long long)(size_t)gptr;
  v4u g0;
  g0.x = 1u;                                     // count=1, user descriptor
  g0.y = lds_addr;                               // LDS byte address
  g0.z = (unsigned)ga;                           // global_addr[31:0]
  g0.w = (unsigned)((ga >> 32) & 0x1FFFFFFull)   // global_addr[56:32]
         | (2u << 30);                           // type=2 ("image")
  v8i g1;
  g1[0] = 0x00010000;                            // data_size=1 (2 bytes)
  g1[1] = (int)((tensor_d0 & 0xffffu) << 16);    // tensor_dim0[15:0]
  g1[2] = (int)((tensor_d0 >> 16) | ((tensor_d1 & 0xffffu) << 16));
  g1[3] = (int)((tensor_d1 >> 16) | (tile_d0 << 16));   // | tile_dim0
  g1[4] = (int)(tile_d1 & 0xffffu);              // tile_dim1, tile_dim2=0
  g1[5] = (int)(unsigned)(stride0 & 0xffffffffull);     // dim0_stride lo
  g1[6] = (int)(unsigned)(stride0 >> 32);        // dim0_stride hi, dim1_stride=0
  g1[7] = 0;
  v4i z4 = {0, 0, 0, 0};
  v8i z8 = {0, 0, 0, 0, 0, 0, 0, 0};
  __builtin_amdgcn_tensor_load_to_lds(g0, g1, z4, z4, z8, 0);
}

// ---------------------------------------------------------------- pack X -> f16
__global__ void pack_x(const float* __restrict__ X, _Float16* __restrict__ Xh) {
  int i = (blockIdx.x * blockDim.x + threadIdx.x) * 4;
  float4 v = *(const float4*)(X + i);
  _Float16 h[4] = {(_Float16)v.x, (_Float16)v.y, (_Float16)v.z, (_Float16)v.w};
  *(uint2*)(Xh + i) = *(uint2*)h;
}

// ------------------------------------------------- pack W{q,k,v} -> B-panel f16
// Panel (nt,kt) = 32x16 tile, stored as 32 lanes x 32 bytes, lane l dword j =
// { W[kt*32 + koff + 2j][n], W[kt*32 + koff + 2j+1][n] }, koff=(l<16?0:16),
// n = nt*16 + (l&15).  Contiguous k order, same as the A loads in qkv_gemm.
__global__ void pack_w(const float* __restrict__ Wq, const float* __restrict__ Wk,
                       const float* __restrict__ Wv, _Float16* __restrict__ Wp) {
  int tid = blockIdx.x * blockDim.x + threadIdx.x;   // one dword each
  int j  = tid & 7;
  int l  = (tid >> 3) & 31;
  int kt = (tid >> 8) & 31;
  int nt = tid >> 13;                                // 0..191
  int koff = (l < 16) ? 0 : 16;
  int k  = kt * 32 + koff + 2 * j;
  int ng = nt * 16 + (l & 15);
  const float* W = (ng < 1024) ? Wq : (ng < 2048 ? Wk : Wv);
  int n = ng & 1023;
  _Float16 h2[2] = {(_Float16)W[k * HID + n], (_Float16)W[(k + 1) * HID + n]};
  ((unsigned*)Wp)[tid] = *(unsigned*)h2;
}

// ------------------------------------------------------------- fused QKV GEMM
// one wave -> 32 rows x 64 cols (8 accumulators), register double-buffered
__global__ __launch_bounds__(32) void qkv_gemm(
    const _Float16* __restrict__ Xh, const _Float16* __restrict__ Wp,
    const float* __restrict__ bq, const float* __restrict__ bk,
    const float* __restrict__ bv,
    _Float16* __restrict__ Qh, _Float16* __restrict__ Kh,
    _Float16* __restrict__ Vt) {
  const int l   = threadIdx.x;
  const int mt  = blockIdx.x & 127;   // 128 row tiles of 32
  const int nt4 = blockIdx.x >> 7;    // 48 col strips of 64
  const int m    = l & 15;
  const int aoff = (l < 16) ? 0 : 16;

  const _Float16* arow0 = Xh + (size_t)(mt * 32 + m) * HID;
  const _Float16* arow1 = arow0 + (size_t)16 * HID;

  v8f c[8] = {v8f{}, v8f{}, v8f{}, v8f{}, v8f{}, v8f{}, v8f{}, v8f{}};

  auto ldA = [&](const _Float16* base, int kt) {
    const _Float16* p = base + kt * 32 + aoff;
    return load_frag2(p, p + 8);
  };
  auto ldB = [&](int i, int kt) {
    const _Float16* pb = Wp + (size_t)((nt4 * 4 + i) * 32 + kt) * 512 + l * 16;
    return load_frag2(pb, pb + 8);
  };

  v16h a0 = ldA(arow0, 0), a1 = ldA(arow1, 0);
  v16h b0 = ldB(0, 0), b1 = ldB(1, 0), b2 = ldB(2, 0), b3 = ldB(3, 0);

  for (int kt = 1; kt < 32; ++kt) {
    v16h na0 = ldA(arow0, kt), na1 = ldA(arow1, kt);
    v16h nb0 = ldB(0, kt), nb1 = ldB(1, kt), nb2 = ldB(2, kt), nb3 = ldB(3, kt);

    c[0] = wmma32(a0, b0, c[0]); c[1] = wmma32(a0, b1, c[1]);
    c[2] = wmma32(a0, b2, c[2]); c[3] = wmma32(a0, b3, c[3]);
    c[4] = wmma32(a1, b0, c[4]); c[5] = wmma32(a1, b1, c[5]);
    c[6] = wmma32(a1, b2, c[6]); c[7] = wmma32(a1, b3, c[7]);

    a0 = na0; a1 = na1; b0 = nb0; b1 = nb1; b2 = nb2; b3 = nb3;
  }
  // drain last k-step
  c[0] = wmma32(a0, b0, c[0]); c[1] = wmma32(a0, b1, c[1]);
  c[2] = wmma32(a0, b2, c[2]); c[3] = wmma32(a0, b3, c[3]);
  c[4] = wmma32(a1, b0, c[4]); c[5] = wmma32(a1, b1, c[5]);
  c[6] = wmma32(a1, b2, c[6]); c[7] = wmma32(a1, b3, c[7]);

  // epilogue: bias, Q scale, scatter into head-major / transposed-V layouts
#pragma unroll
  for (int half = 0; half < 2; ++half) {
#pragma unroll
    for (int i = 0; i < 4; ++i) {
      int ng  = nt4 * 64 + i * 16 + (l & 15);
      int mat = ng >> 10;
      int nl  = ng & 1023;
      int h = nl >> 6, d = nl & 63;
      const float* bias = (mat == 0) ? bq : (mat == 1 ? bk : bv);
      float bb = bias[nl];
      const v8f& acc = c[half * 4 + i];
#pragma unroll
      for (int r = 0; r < 8; ++r) {
        int row = mt * 32 + half * 16 + r + ((l >> 4) << 3);
        int b_  = row >> 11, s = row & 2047;
        float val = acc[r] + bb;
        if (mat == 0)
          Qh[((size_t)(b_ * NHEAD + h) * SEQ + s) * HDIM + d] = (_Float16)(val * 0.125f);
        else if (mat == 1)
          Kh[((size_t)(b_ * NHEAD + h) * SEQ + s) * HDIM + d] = (_Float16)val;
        else
          Vt[((size_t)(b_ * NHEAD + h) * HDIM + d) * SEQ + s] = (_Float16)val;
      }
    }
  }
}

// ------------------------------------------------------- flash attention block
// block = 4 waves = 64 query rows of one (b,h).  K tile streams via the Tensor
// Data Mover (TENSORcnt), V^T tile via async global->LDS (ASYNCcnt); both
// double-buffered in LDS and shared by all 4 waves.
__global__ __launch_bounds__(128) void attention(
    const _Float16* __restrict__ Qh, const _Float16* __restrict__ Kh,
    const _Float16* __restrict__ Vt, const float* __restrict__ mask,
    float* __restrict__ out) {
  __shared__ __align__(16) _Float16 ldsK[2][32 * 64];   // [key][d]   4 KB x2
  __shared__ __align__(16) _Float16 ldsV[2][64 * 32];   // [d][key]   4 KB x2
  __shared__ __align__(16) _Float16 ldsP[4][16 * 32];   // per-wave P 1 KB x4

  const int tid  = threadIdx.x;
  const int wave = tid >> 5;
  const int l    = tid & 31;

  const int qt64 = blockIdx.x & 31;    // 32 tiles of 64 q rows
  const int bh   = blockIdx.x >> 5;    // 32 (b,h) pairs
  const int b_ = bh >> 4;
  const int h_ = bh & 15;
  const int q0 = qt64 * 64 + wave * 16;

  const _Float16* Qbh = Qh + (size_t)bh * SEQ * HDIM;
  const _Float16* Kbh = Kh + (size_t)bh * SEQ * HDIM;
  const _Float16* Vbh = Vt + (size_t)bh * HDIM * SEQ;
  const float*    msk = mask + b_ * SEQ;

  const int m    = l & 15;
  const int aoff = (l < 16) ? 0 : 16;
  const int koff = (l < 16) ? 0 : 16;

  // Q A-fragments for d=[0,32) and d=[32,64)
  const _Float16* qp = Qbh + (size_t)(q0 + m) * HDIM;
  v16h qa0 = load_frag2(qp + aoff,      qp + 8 + aoff);
  v16h qa1 = load_frag2(qp + 32 + aoff, qp + 40 + aoff);

  // stage one 32-key tile: K via TDM (contiguous 32x64 rows),
  // V^T via async copies (64 rows x 64B strided by SEQ)
  auto issue_tile = [&](int buf, int key0) {
    tdm_load_2d((unsigned)(size_t)&ldsK[buf][0],
                Kbh + (size_t)key0 * HDIM,
                /*tensor_d0=*/HDIM, /*tensor_d1=*/SEQ,
                /*tile_d0=*/HDIM, /*tile_d1=*/32, /*stride0=*/HDIM);
    unsigned vbase = (unsigned)(size_t)&ldsV[buf][0];
#pragma unroll
    for (int i = 0; i < 8; ++i) {
      int chunk = i * 32 + l;              // 0..255, 16B each
      int row = chunk >> 2, sub = chunk & 3;
      const char* vg = (const char*)Vbh + ((size_t)row * SEQ + key0) * 2 + sub * 16;
      async_copy16(vbase + (unsigned)chunk * 16, vg);
    }
  };

  if (wave == 0) issue_tile(0, 0);

  v8f o[4] = {v8f{}, v8f{}, v8f{}, v8f{}};
  float mprev[8], lsum[8];
#pragma unroll
  for (int r = 0; r < 8; ++r) { mprev[r] = -3.0e38f; lsum[r] = 0.f; }

  for (int key0 = 0; key0 < SEQ; key0 += 32) {
    const int buf = (key0 >> 5) & 1;
    if (wave == 0) {
      if (key0 + 32 < SEQ) {
        issue_tile(buf ^ 1, key0 + 32);
        __builtin_amdgcn_s_wait_tensorcnt(1);   // previous K tile landed
        wait_async_le8();                       // previous V tile landed
      } else {
        __builtin_amdgcn_s_wait_tensorcnt(0);
        wait_async_0();
      }
    }
    __syncthreads();                       // tile[buf] visible to all waves

    // K^T B-fragments from LDS (ds_load_b128 pairs)
    const _Float16* kt0 = &ldsK[buf][0];
    const _Float16* p0 = kt0 + (size_t)(l & 15) * HDIM + koff;
    const _Float16* p1 = p0 + 16 * HDIM;
    v16h kb00 = load_frag2(p0, p0 + 8);
    v16h kb01 = load_frag2(p0 + 32, p0 + 40);
    v16h kb10 = load_frag2(p1, p1 + 8);
    v16h kb11 = load_frag2(p1 + 32, p1 + 40);

    v8f s0 = {}, s1 = {};
    s0 = wmma32(qa0, kb00, s0); s0 = wmma32(qa1, kb01, s0);
    s1 = wmma32(qa0, kb10, s1); s1 = wmma32(qa1, kb11, s1);

    float mv0 = msk[key0 + (l & 15)];
    float mv1 = msk[key0 + 16 + (l & 15)];

    // online softmax (rows live in half-wave lane groups; xor<=8 stays inside)
    float corr[8];
#pragma unroll
    for (int r = 0; r < 8; ++r) {
      float a0 = s0[r] + mv0;
      float a1 = s1[r] + mv1;
      float tmax = fmaxf(a0, a1);
#pragma unroll
      for (int sh = 8; sh >= 1; sh >>= 1)
        tmax = fmaxf(tmax, __shfl_xor(tmax, sh, 32));
      float mnew = fmaxf(mprev[r], tmax);
      float e0 = __expf(a0 - mnew);
      float e1 = __expf(a1 - mnew);
      float rs = e0 + e1;
#pragma unroll
      for (int sh = 8; sh >= 1; sh >>= 1)
        rs += __shfl_xor(rs, sh, 32);
      corr[r]  = __expf(mprev[r] - mnew);
      lsum[r]  = lsum[r] * corr[r] + rs;
      mprev[r] = mnew;
      s0[r] = e0; s1[r] = e1;
    }
#pragma unroll
    for (int r = 0; r < 8; ++r) {
      o[0][r] *= corr[r]; o[1][r] *= corr[r];
      o[2][r] *= corr[r]; o[3][r] *= corr[r];
    }

    // transpose P (C layout -> A layout) through per-wave LDS
    {
      int prow = (l >> 4) << 3;
      int pc   = l & 15;
#pragma unroll
      for (int r = 0; r < 8; ++r) {
        ldsP[wave][(prow + r) * 32 + pc]      = (_Float16)s0[r];
        ldsP[wave][(prow + r) * 32 + 16 + pc] = (_Float16)s1[r];
      }
    }
    const _Float16* pp = &ldsP[wave][0] + m * 32 + aoff;
    v16h pa = load_frag2(pp, pp + 8);

    // PV: V^T tile rows are contiguous along keys in LDS
    const _Float16* vt0 = &ldsV[buf][0];
#pragma unroll
    for (int i = 0; i < 4; ++i) {
      const _Float16* vp = vt0 + (size_t)(i * 16 + (l & 15)) * 32 + koff;
      v16h vb = load_frag2(vp, vp + 8);
      o[i] = wmma32(pa, vb, o[i]);
    }
    __syncthreads();                       // all waves done with tile[buf]
  }

  // normalize and write f32 output [B,S,HID]
  const int dl = l & 15;
#pragma unroll
  for (int r = 0; r < 8; ++r) {
    int srow  = q0 + r + ((l >> 4) << 3);
    float inv = 1.0f / lsum[r];
    float* orow = out + (size_t)(b_ * SEQ + srow) * HID + h_ * HDIM + dl;
    orow[0]  = o[0][r] * inv;
    orow[16] = o[1][r] * inv;
    orow[32] = o[2][r] * inv;
    orow[48] = o[3][r] * inv;
  }
}

extern "C" void kernel_launch(void* const* d_in, const int* in_sizes, int n_in,
                              void* d_out, int out_size, void* d_ws, size_t ws_size,
                              hipStream_t stream) {
  const float* X   = (const float*)d_in[0];
  const float* msk = (const float*)d_in[1];
  const float* Wq  = (const float*)d_in[2];
  const float* bq  = (const float*)d_in[3];
  const float* Wk  = (const float*)d_in[4];
  const float* bk  = (const float*)d_in[5];
  const float* Wv  = (const float*)d_in[6];
  const float* bv  = (const float*)d_in[7];
  float* out = (float*)d_out;

  char* ws = (char*)d_ws;
  _Float16* Xh = (_Float16*)(ws + OFF_XH);
  _Float16* Wp = (_Float16*)(ws + OFF_WP);
  _Float16* Qh = (_Float16*)(ws + OFF_QH);
  _Float16* Kh = (_Float16*)(ws + OFF_KH);
  _Float16* Vt = (_Float16*)(ws + OFF_VT);

  pack_x<<<TOKENS * HID / (256 * 4), 256, 0, stream>>>(X, Xh);
  pack_w<<<(HID * N3 / 2) / 256, 256, 0, stream>>>(Wq, Wk, Wv, Wp);
  qkv_gemm<<<128 * (N3 / 64), 32, 0, stream>>>(Xh, Wp, bq, bk, bv, Qh, Kh, Vt);
  attention<<<32 * (BATCH * NHEAD), 128, 0, stream>>>(Qh, Kh, Vt, msk, out);
}
